// SegformerWithRoPEEfficientSelfAttention_71124658422000
// MI455X (gfx1250) — matrix operand order repaired and checked by reading, not compiled, not run on hardware
//
#include <hip/hip_runtime.h>
#include <hip/hip_bf16.h>

typedef _Float16 half_t;
typedef __attribute__((ext_vector_type(16))) _Float16 v16h;
typedef __attribute__((ext_vector_type(8)))  _Float16 v8h;
typedef __attribute__((ext_vector_type(8)))  float    v8f;

#define B_   2
#define S_   16384
#define C_   128
#define H_   2
#define D_   64
#define ROT_ 32      // D_/2
#define SK_  1024
#define HW_  128     // height == width == 128 (fixed by setup_inputs)
#define OW_  32      // HW_/SR
#define SRr  4
#define TS_  16

#define WMMA_F16(A, B, C) \
  __builtin_amdgcn_wmma_f32_16x16x32_f16(false, (A), false, (B), (short)0, (C), false, false)

// ---------------------------------------------------------------------------
// K1: q = hidden @ Wq.T + bq, then RoPE, store f16 as [b][h][s][d]
// ---------------------------------------------------------------------------
__global__ void qproj_rope_kernel(const float* __restrict__ hidden,
                                  const float* __restrict__ Wq,
                                  const float* __restrict__ bq,
                                  const float* __restrict__ freqs,
                                  half_t* __restrict__ q16) {
  __shared__ float hrows[TS_][C_];
  __shared__ float qrows[TS_][C_];
  const int blk = blockIdx.x;               // over B*S/TS
  const int b   = blk / (S_ / TS_);
  const int s0  = (blk % (S_ / TS_)) * TS_;
  const int t   = threadIdx.x;              // 128 threads: one output channel each

  for (int r = 0; r < TS_; ++r)
    hrows[r][t] = hidden[(((size_t)b * S_) + s0 + r) * C_ + t];
  __syncthreads();

  const float* w = Wq + (size_t)t * C_;
  const float bias = bq[t];
  float acc[TS_];
#pragma unroll
  for (int r = 0; r < TS_; ++r) acc[r] = bias;

  for (int k = 0; k < C_; k += 4) {
    float4 w4 = *(const float4*)(w + k);
#pragma unroll
    for (int r = 0; r < TS_; ++r) {
      float4 h4 = *(const float4*)(&hrows[r][k]);
      acc[r] += h4.x * w4.x + h4.y * w4.y + h4.z * w4.z + h4.w * w4.w;
    }
  }
#pragma unroll
  for (int r = 0; r < TS_; ++r) qrows[r][t] = acc[r];
  __syncthreads();

  // RoPE: 16 rows x 64 pairs = 1024 tasks
  for (int task = t; task < TS_ * 64; task += 128) {
    const int r  = task >> 6;
    const int u  = task & 63;
    const int hh = u >> 5;
    const int j  = u & 31;
    const int s  = s0 + r;
    const float xr = qrows[r][hh * D_ + 2 * j];
    const float xi = qrows[r][hh * D_ + 2 * j + 1];
    const float fr = freqs[((size_t)s * ROT_ + j) * 2 + 0];
    const float fi = freqs[((size_t)s * ROT_ + j) * 2 + 1];
    const size_t base = (((size_t)(b * H_ + hh)) * S_ + s) * D_;
    q16[base + 2 * j]     = (half_t)(xr * fr - xi * fi);
    q16[base + 2 * j + 1] = (half_t)(xr * fi + xi * fr);
  }
}

// ---------------------------------------------------------------------------
// K2: 4x4/stride-4 conv (spatial reduction) + bias + LayerNorm -> xr f32
// ---------------------------------------------------------------------------
__global__ void sr_conv_ln_kernel(const float* __restrict__ hidden,
                                  const float* __restrict__ Wsr,
                                  const float* __restrict__ bsr,
                                  const float* __restrict__ lng,
                                  const float* __restrict__ lnb,
                                  float* __restrict__ xr) {
  __shared__ float patch[16 * C_];
  __shared__ float red[C_];
  const int blk = blockIdx.x;          // b*SK + sk
  const int b   = blk / SK_;
  const int sk  = blk % SK_;
  const int oy  = sk / OW_, ox = sk % OW_;
  const int t   = threadIdx.x;

  for (int i = t; i < 16 * C_; i += C_) {
    const int p  = i >> 7;
    const int ci = i & 127;
    const int ky = p >> 2, kx = p & 3;
    const int s2 = (oy * SRr + ky) * HW_ + ox * SRr + kx;
    patch[i] = hidden[(((size_t)b * S_) + s2) * C_ + ci];
  }
  __syncthreads();

  const float* w = Wsr + (size_t)t * C_ * 16;   // W_sr[t][ci][ky][kx]
  float acc = bsr[t];
  for (int ci = 0; ci < C_; ++ci) {
    const float* wp = w + ci * 16;
#pragma unroll
    for (int p = 0; p < 16; ++p) acc += patch[p * C_ + ci] * wp[p];
  }

  // LayerNorm over channels
  red[t] = acc;
  __syncthreads();
  for (int sft = 64; sft > 0; sft >>= 1) {
    if (t < sft) red[t] += red[t + sft];
    __syncthreads();
  }
  const float mu = red[0] * (1.0f / C_);
  __syncthreads();
  const float diff = acc - mu;
  red[t] = diff * diff;
  __syncthreads();
  for (int sft = 64; sft > 0; sft >>= 1) {
    if (t < sft) red[t] += red[t + sft];
    __syncthreads();
  }
  const float var = red[0] * (1.0f / C_);
  xr[(size_t)blk * C_ + t] = diff * rsqrtf(var + 1e-5f) * lng[t] + lnb[t];
}

// ---------------------------------------------------------------------------
// K3: k/v = xr @ Wk.T/Wv.T + b, RoPE(k) with patch-averaged freqs.
//     k f16 [b][h][sk][d]; v f16 TRANSPOSED [b][h][d][sk]
// ---------------------------------------------------------------------------
__global__ void kv_proj_rope_kernel(const float* __restrict__ xr,
                                    const float* __restrict__ Wk,
                                    const float* __restrict__ bk,
                                    const float* __restrict__ Wv,
                                    const float* __restrict__ bv,
                                    const float* __restrict__ freqs,
                                    half_t* __restrict__ k16,
                                    half_t* __restrict__ vT16) {
  __shared__ float xrow[C_];
  __shared__ float krow[C_];
  __shared__ float fravg[ROT_], fiavg[ROT_];
  const int blk = blockIdx.x;          // b*SK + sk
  const int b   = blk / SK_;
  const int sk  = blk % SK_;
  const int oy  = sk / OW_, ox = sk % OW_;
  const int t   = threadIdx.x;

  xrow[t] = xr[(size_t)blk * C_ + t];
  if (t < 64) {
    const int which = t >> 5;
    const int j = t & 31;
    float sum = 0.f;
#pragma unroll
    for (int p = 0; p < 16; ++p) {
      const int s2 = (oy * SRr + (p >> 2)) * HW_ + ox * SRr + (p & 3);
      sum += freqs[((size_t)s2 * ROT_ + j) * 2 + which];
    }
    sum *= (1.0f / 16.0f);
    if (which == 0) fravg[j] = sum; else fiavg[j] = sum;
  }
  __syncthreads();

  const float* wk = Wk + (size_t)t * C_;
  const float* wv = Wv + (size_t)t * C_;
  float ka = bk[t], va = bv[t];
  for (int k = 0; k < C_; k += 4) {
    float4 x4 = *(const float4*)(&xrow[k]);
    float4 k4 = *(const float4*)(wk + k);
    float4 v4 = *(const float4*)(wv + k);
    ka += x4.x * k4.x + x4.y * k4.y + x4.z * k4.z + x4.w * k4.w;
    va += x4.x * v4.x + x4.y * v4.y + x4.z * v4.z + x4.w * v4.w;
  }
  krow[t] = ka;
  {
    const int hh = t >> 6;
    const int dc = t & 63;
    vT16[(((size_t)(b * H_ + hh)) * D_ + dc) * SK_ + sk] = (half_t)va;
  }
  __syncthreads();
  if (t < 64) {
    const int hh = t >> 5;
    const int j = t & 31;
    const float xrv = krow[hh * D_ + 2 * j];
    const float xiv = krow[hh * D_ + 2 * j + 1];
    const float fr = fravg[j], fi = fiavg[j];
    const size_t base = (((size_t)(b * H_ + hh)) * SK_ + sk) * D_;
    k16[base + 2 * j]     = (half_t)(xrv * fr - xiv * fi);
    k16[base + 2 * j + 1] = (half_t)(xrv * fi + xiv * fr);
  }
}

// ---------------------------------------------------------------------------
// K4: flash attention. 1 wave per 16-row q tile, 4 waves per block.
//     Per 32-col K-step: 4 score WMMAs + online softmax (LDS bounce) + 4 PV WMMAs
// ---------------------------------------------------------------------------
__global__ void attn_kernel(const half_t* __restrict__ q16,
                            const half_t* __restrict__ k16,
                            const half_t* __restrict__ vT16,
                            float* __restrict__ out) {
  __shared__ float s_tile[4][16][32];
  __shared__ __align__(32) half_t p_tile[4][16][32];
  __shared__ float s_m[4][16];
  __shared__ float s_l[4][16];
  __shared__ float s_sc[4][16];

  const int tid  = threadIdx.x;
  const int w    = tid >> 5;
  const int lane = tid & 31;
  const int n    = lane & 15;   // N column / A row within tile
  const int hi   = lane >> 4;

  const int tileId = blockIdx.x * 4 + w;          // [0, 4096)
  const int b  = tileId >> 11;
  const int h  = (tileId >> 10) & 1;
  const int qt = tileId & 1023;

  // q A-fragments (16x32 f16 layout: lane m=L%16; K = hi*16 + j)
  const half_t* qrow = q16 + (((size_t)(b * H_ + h)) * S_ + (size_t)qt * 16 + n) * D_;
  v16h aq0, aq1;
  {
    v8h x0 = *(const v8h*)(qrow + hi * 16);
    v8h x1 = *(const v8h*)(qrow + 8 + hi * 16);
    v8h x2 = *(const v8h*)(qrow + 32 + hi * 16);
    v8h x3 = *(const v8h*)(qrow + 40 + hi * 16);
#pragma unroll
    for (int i = 0; i < 8; ++i) {
      aq0[i] = x0[i]; aq0[8 + i] = x1[i];
      aq1[i] = x2[i]; aq1[8 + i] = x3[i];
    }
  }

  v8f o0 = {0,0,0,0,0,0,0,0}, o1 = {0,0,0,0,0,0,0,0};
  v8f o2 = {0,0,0,0,0,0,0,0}, o3 = {0,0,0,0,0,0,0,0};
  if (lane < 16) { s_m[w][lane] = -3.0e38f; s_l[w][lane] = 0.f; }
  __syncthreads();

  const half_t* kbase = k16  + ((size_t)(b * H_ + h)) * SK_ * D_;
  const half_t* vbase = vT16 + ((size_t)(b * H_ + h)) * D_ * SK_;
  const float sc = 0.125f;   // 1/sqrt(64)

  for (int it = 0; it < SK_ / 32; ++it) {
    // ---- scores: two 16x16 tiles over K=d=64 ----
    const half_t* kr0 = kbase + ((size_t)(it * 32) + n) * D_;
    const half_t* kr1 = kr0 + 16 * D_;
    v16h bk00 = *(const v16h*)(kr0 + hi * 16);
    v16h bk01 = *(const v16h*)(kr0 + 32 + hi * 16);
    v16h bk10 = *(const v16h*)(kr1 + hi * 16);
    v16h bk11 = *(const v16h*)(kr1 + 32 + hi * 16);
    v8f sA = {0,0,0,0,0,0,0,0}, sB = {0,0,0,0,0,0,0,0};
    sA = WMMA_F16(aq0, bk00, sA);
    sA = WMMA_F16(aq1, bk01, sA);
    sB = WMMA_F16(aq0, bk10, sB);
    sB = WMMA_F16(aq1, bk11, sB);
#pragma unroll
    for (int r = 0; r < 8; ++r) {
      s_tile[w][hi * 8 + r][n]      = sA[r] * sc;
      s_tile[w][hi * 8 + r][n + 16] = sB[r] * sc;
    }
    __syncthreads();

    // ---- online softmax update (one lane per row) ----
    if (lane < 16) {
      const float mo = s_m[w][lane];
      float tmax = mo;
#pragma unroll
      for (int cc = 0; cc < 32; ++cc) tmax = fmaxf(tmax, s_tile[w][lane][cc]);
      const float esc = __expf(mo - tmax);
      float lsum = s_l[w][lane] * esc;
#pragma unroll
      for (int cc = 0; cc < 32; ++cc) {
        const float p = __expf(s_tile[w][lane][cc] - tmax);
        lsum += p;
        p_tile[w][lane][cc] = (half_t)p;
      }
      s_m[w][lane] = tmax;
      s_l[w][lane] = lsum;
      s_sc[w][lane] = esc;
    }
    __syncthreads();

    // ---- rescale accumulators ----
    float rs[8];
#pragma unroll
    for (int r = 0; r < 8; ++r) rs[r] = s_sc[w][hi * 8 + r];
#pragma unroll
    for (int r = 0; r < 8; ++r) {
      o0[r] *= rs[r]; o1[r] *= rs[r]; o2[r] *= rs[r]; o3[r] *= rs[r];
    }

    // ---- P A-fragment from LDS (row-major 16x32) ----
    v16h ap;
    {
      v8h p0 = *(const v8h*)(&p_tile[w][n][hi * 16]);
      v8h p1 = *(const v8h*)(&p_tile[w][n][8 + hi * 16]);
#pragma unroll
      for (int i = 0; i < 8; ++i) { ap[i] = p0[i]; ap[8 + i] = p1[i]; }
    }

    // ---- V B-fragments from vT (K contiguous) + 4 PV WMMAs ----
    const half_t* vr = vbase + (size_t)n * SK_ + it * 32 + hi * 16;
    v16h bv0 = *(const v16h*)(vr);
    v16h bv1 = *(const v16h*)(vr + 16 * SK_);
    v16h bv2 = *(const v16h*)(vr + 32 * SK_);
    v16h bv3 = *(const v16h*)(vr + 48 * SK_);
    o0 = WMMA_F16(ap, bv0, o0);
    o1 = WMMA_F16(ap, bv1, o1);
    o2 = WMMA_F16(ap, bv2, o2);
    o3 = WMMA_F16(ap, bv3, o3);
    __syncthreads();
  }

  // ---- normalize and store (out[b][s][h*64 + d]) ----
  float invl[8];
#pragma unroll
  for (int r = 0; r < 8; ++r) invl[r] = 1.0f / s_l[w][hi * 8 + r];
  const size_t ob = ((size_t)b * S_ + (size_t)qt * 16) * C_ + h * D_ + n;
#pragma unroll
  for (int r = 0; r < 8; ++r) {
    const size_t rowoff = ob + (size_t)(hi * 8 + r) * C_;
    out[rowoff]      = o0[r] * invl[r];
    out[rowoff + 16] = o1[r] * invl[r];
    out[rowoff + 32] = o2[r] * invl[r];
    out[rowoff + 48] = o3[r] * invl[r];
  }
}

// ---------------------------------------------------------------------------
extern "C" void kernel_launch(void* const* d_in, const int* in_sizes, int n_in,
                              void* d_out, int out_size, void* d_ws, size_t ws_size,
                              hipStream_t stream) {
  const float* hidden = (const float*)d_in[0];
  const float* freqs  = (const float*)d_in[1];
  const float* Wq  = (const float*)d_in[2];
  const float* bq  = (const float*)d_in[3];
  const float* Wk  = (const float*)d_in[4];
  const float* bk  = (const float*)d_in[5];
  const float* Wv  = (const float*)d_in[6];
  const float* bv  = (const float*)d_in[7];
  const float* Wsr = (const float*)d_in[8];
  const float* bsr = (const float*)d_in[9];
  const float* lng = (const float*)d_in[10];
  const float* lnb = (const float*)d_in[11];
  (void)in_sizes; (void)n_in; (void)out_size; (void)ws_size;
  // height/width fixed at 128 by setup_inputs

  char* p = (char*)d_ws;
  half_t* qh = (half_t*)p;  p += (size_t)B_ * H_ * S_  * D_ * sizeof(half_t);  // 8 MB
  half_t* kh = (half_t*)p;  p += (size_t)B_ * H_ * SK_ * D_ * sizeof(half_t);  // 512 KB
  half_t* vT = (half_t*)p;  p += (size_t)B_ * H_ * D_ * SK_ * sizeof(half_t);  // 512 KB
  float*  xrb = (float*)p;  p += (size_t)B_ * SK_ * C_ * sizeof(float);        // 1 MB

  hipLaunchKernelGGL(qproj_rope_kernel, dim3(B_ * S_ / TS_), dim3(128), 0, stream,
                     hidden, Wq, bq, freqs, qh);
  hipLaunchKernelGGL(sr_conv_ln_kernel, dim3(B_ * SK_), dim3(128), 0, stream,
                     hidden, Wsr, bsr, lng, lnb, xrb);
  hipLaunchKernelGGL(kv_proj_rope_kernel, dim3(B_ * SK_), dim3(128), 0, stream,
                     xrb, Wk, bk, Wv, bv, freqs, kh, vT);
  hipLaunchKernelGGL(attn_kernel, dim3(B_ * H_ * (S_ / 16) / 4), dim3(128), 0, stream,
                     qh, kh, vT, (float*)d_out);
}
